// EpipolarCrossAttention_15083925143761
// MI455X (gfx1250) — matrix operand order repaired and checked by reading, not compiled, not run on hardware
//
#include <hip/hip_runtime.h>
#include <hip/hip_bf16.h>
#include <math.h>

// ---------------------------------------------------------------------------
// EpipolarCrossAttention on MI455X (gfx1250, wave32, WMMA bf16 16x16x32)
//
// Algebraic restructuring (single query per (b,q) row, K=160 keys):
//   scores = X @ r           r[c,h] = sum_d Wk[c,32h+d] * qproj[32h+d] / sqrt(32)
//   u[h,:] = softmax(scores)[:,h]^T @ X
//   y      = U_flat(2880x2048) @ Mstack(2048x256) + c1,  Mstack = blockdiag fold of Wv,Wo
//   out    = x + FFN(LN(x)),  x = target + y
// sampled_features (472 MB) is read once from HBM (plus an L2-hot re-read),
// making the block HBM-bound (~20 us floor at 23.3 TB/s). All matmuls use
// v_wmma_f32_16x16x32_bf16; per-row r is staged to LDS with the CDNA5 async
// LDS mover (ASYNCcnt) when the toolchain exposes it.
// ---------------------------------------------------------------------------

#define DEV __device__ __forceinline__

typedef __attribute__((ext_vector_type(16))) __bf16 v16bf;
typedef __attribute__((ext_vector_type(8)))  float  v8f;
typedef int v4i __attribute__((vector_size(16)));   // matches builtin's V4i

union BF16Tile { uint4 u4[2]; v16bf v; };
union Acc8     { v8f v; float f[8]; };

static DEV void zero8(Acc8& a) {
#pragma unroll
  for (int i = 0; i < 8; ++i) a.f[i] = 0.0f;
}

static DEV unsigned short f2bf(float x) {            // RNE float -> bf16
  unsigned u = __float_as_uint(x);
  unsigned r = u + 0x7FFFu + ((u >> 16) & 1u);
  return (unsigned short)(r >> 16);
}
static DEV unsigned pack2(float a, float b) {
  return (unsigned)f2bf(a) | ((unsigned)f2bf(b) << 16);
}

static DEV v8f wmma_bf16(const BF16Tile& a, const BF16Tile& b, v8f c) {
  // D = A(16x32 bf16) x B(32x16 bf16) + C(16x16 f32)
  return __builtin_amdgcn_wmma_f32_16x16x32_bf16(false, a.v, false, b.v,
                                                 (short)0, c, false, false);
}

// ---- CDNA5 async global->LDS copy (ASYNCcnt) with compile-safe fallback ----
#if __has_builtin(__builtin_amdgcn_global_load_async_to_lds_b128) && \
    __has_builtin(__builtin_amdgcn_s_wait_asynccnt)
#define USE_ASYNC_LDS 1
#else
#define USE_ASYNC_LDS 0
#endif

static DEV void copy16_to_lds(const void* g, void* l) {
#if USE_ASYNC_LDS
  __builtin_amdgcn_global_load_async_to_lds_b128(
      (__attribute__((address_space(1))) v4i*)g,
      (__attribute__((address_space(3))) v4i*)l, 0, 0);
#else
  *(uint4*)l = *(const uint4*)g;
#endif
}
static DEV void async_lds_wait() {
#if USE_ASYNC_LDS
  __builtin_amdgcn_s_wait_asynccnt(0);
#endif
}

// A-operand per lane: rows m=lane&15; comps: k = k0 + half*16 + c
//   -> two contiguous uint4 at (row*lda + half*16 + k0) and (+8)
// B-operand per lane: cols n=lane&15; comps: k = k0 + half*16 + c
//   -> two contiguous uint4 at (n*ldb + half*16 + k0) and (+8)

static constexpr int NR = 2880;   // B*Q rows
static constexpr int QL = 720;
static constexpr int KK = 160;    // keys per row
static constexpr int DM = 256;

// ---------------------------------------------------------------- prep kernels
__global__ void __launch_bounds__(256)
k_convert(const float* __restrict__ src, unsigned short* __restrict__ dst,
          int R, int C, int trans) {
  int idx = blockIdx.x * 256 + threadIdx.x;
  if (idx >= R * C) return;
  int r = idx / C, c = idx - r * C;
  unsigned short v = f2bf(src[idx]);
  if (trans) dst[(size_t)c * R + r] = v;   // dst = src^T (C x R row-major)
  else       dst[idx] = v;
}

// MstackT[o][h*256+c] = sum_d Wv[c][32h+d] * Wo[32h+d][o]   (bf16, B-operand layout)
__global__ void __launch_bounds__(256)
k_mstack(const float* __restrict__ Wv, const float* __restrict__ Wo,
         unsigned short* __restrict__ MT) {
  int idx = blockIdx.x * 256 + threadIdx.x;    // 256*2048 outputs
  int o = idx >> 11, t = idx & 2047;
  int h = t >> 8,    c = t & 255;
  const float* wv = Wv + (size_t)c * 256 + h * 32;
  const float* wo = Wo + (size_t)(h * 32) * 256 + o;
  float acc = 0.f;
#pragma unroll
  for (int d = 0; d < 32; ++d) acc += wv[d] * wo[(size_t)d * 256];
  MT[(size_t)o * 2048 + t] = f2bf(acc);
}

// c1[o] = sum_j bv[j]*Wo[j][o] + bo[o]
__global__ void __launch_bounds__(256)
k_c1(const float* __restrict__ bv, const float* __restrict__ Wo,
     const float* __restrict__ bo, float* __restrict__ c1) {
  int o = threadIdx.x;
  float acc = bo[o];
  for (int j = 0; j < 256; ++j) acc += bv[j] * Wo[(size_t)j * 256 + o];
  c1[o] = acc;
}

// ------------------------------------------------------------------ layernorm
__global__ void __launch_bounds__(256)
k_layernorm(const float* __restrict__ x, const float* __restrict__ g,
            const float* __restrict__ bb, unsigned short* __restrict__ out) {
  int wave = threadIdx.x >> 5, lane = threadIdx.x & 31;
  int row = blockIdx.x * 8 + wave;
  const float* xr = x + (size_t)row * 256;
  float vals[8]; float s = 0.f, s2 = 0.f;
#pragma unroll
  for (int j = 0; j < 8; ++j) {
    float v = xr[lane + 32 * j]; vals[j] = v; s += v; s2 += v * v;
  }
#pragma unroll
  for (int o = 16; o >= 1; o >>= 1) {
    s += __shfl_xor(s, o, 32); s2 += __shfl_xor(s2, o, 32);
  }
  float mu = s * (1.f / 256.f);
  float var = s2 * (1.f / 256.f) - mu * mu;
  float rs = rsqrtf(var + 1e-5f);
  unsigned short* orow = out + (size_t)row * 256;
#pragma unroll
  for (int j = 0; j < 8; ++j) {
    int c = lane + 32 * j;
    orow[c] = f2bf((vals[j] - mu) * rs * g[c] + bb[c]);
  }
}

// ----------------------------------------------------- generic bf16 WMMA GEMM
// One wave computes TWO adjacent 16x16 N-tiles sharing the A operand
// (3 b128 loads per WMMA instead of 4). A row-major (MxK), BT row-major (NxK).
// mode 0: outH = bf16(acc + bias[n])
// mode 1: outF = acc + bias[n] + extra[m][n]          (f32)
// mode 2: outH = bf16(gelu_exact(acc + bias[n]))
__global__ void __launch_bounds__(256)
k_gemm(const unsigned short* __restrict__ A, int lda,
       const unsigned short* __restrict__ BT, int ldb, int K,
       const float* __restrict__ bias,
       const float* __restrict__ extra, int lde,
       float* __restrict__ outF, unsigned short* __restrict__ outH, int ldo,
       int mode) {
  int wave = threadIdx.x >> 5, lane = threadIdx.x & 31;
  int m0 = blockIdx.x * 16;
  int n0 = (blockIdx.y * 8 + wave) * 32;       // two tiles: n0, n0+16
  int q = lane & 15, hf = lane >> 4;
  const unsigned short* Ab  = A + (size_t)(m0 + q) * lda + hf * 16;
  const unsigned short* Bb0 = BT + (size_t)(n0 + q) * ldb + hf * 16;
  const unsigned short* Bb1 = BT + (size_t)(n0 + 16 + q) * ldb + hf * 16;
  Acc8 acc0, acc1; zero8(acc0); zero8(acc1);
  for (int k0 = 0; k0 < K; k0 += 32) {
    BF16Tile a, b0, b1;
    a.u4[0]  = *(const uint4*)(Ab + k0);
    a.u4[1]  = *(const uint4*)(Ab + k0 + 8);
    b0.u4[0] = *(const uint4*)(Bb0 + k0);
    b0.u4[1] = *(const uint4*)(Bb0 + k0 + 8);
    b1.u4[0] = *(const uint4*)(Bb1 + k0);
    b1.u4[1] = *(const uint4*)(Bb1 + k0 + 8);
    acc0.v = wmma_bf16(a, b0, acc0.v);
    acc1.v = wmma_bf16(a, b1, acc1.v);
  }
#pragma unroll
  for (int t = 0; t < 2; ++t) {
    Acc8& acc = t ? acc1 : acc0;
    int n = n0 + 16 * t + q;
#pragma unroll
    for (int v = 0; v < 8; ++v) {
      int m = m0 + v + 8 * hf;
      float val = acc.f[v] + bias[n];
      if (mode == 1) {
        outF[(size_t)m * ldo + n] = val + extra[(size_t)m * lde + n];
      } else if (mode == 2) {
        val = 0.5f * val * (1.f + erff(val * 0.70710678118654752f));
        outH[(size_t)m * ldo + n] = f2bf(val);
      } else {
        outH[(size_t)m * ldo + n] = f2bf(val);
      }
    }
  }
}

// ---------------------------------------------------- fold: rT = qp_h @ Wk_h^T
// rT[row][h][c] = (1/sqrt(32)) * sum_d qp[row][32h+d] * Wk[c][32h+d]
__global__ void __launch_bounds__(256)
k_fold(const unsigned short* __restrict__ qp, const unsigned short* __restrict__ Wk,
       unsigned short* __restrict__ rT) {
  int wave = threadIdx.x >> 5, lane = threadIdx.x & 31;
  int m0 = blockIdx.x * 16;
  int n0 = (blockIdx.y * 8 + wave) * 16;
  int h  = blockIdx.z;
  int q = lane & 15, hf = lane >> 4;
  const unsigned short* Ab = qp + (size_t)(m0 + q) * 256 + h * 32 + hf * 16;
  const unsigned short* Bb = Wk + (size_t)(n0 + q) * 256 + h * 32 + hf * 16;
  BF16Tile a, b;
  a.u4[0] = *(const uint4*)(Ab);      a.u4[1] = *(const uint4*)(Ab + 8);
  b.u4[0] = *(const uint4*)(Bb);      b.u4[1] = *(const uint4*)(Bb + 8);
  Acc8 acc; zero8(acc);
  acc.v = wmma_bf16(a, b, acc.v);
  const float sc = 0.17677669529663688f;   // 1/sqrt(32)
#pragma unroll
  for (int v = 0; v < 8; ++v) {
    int row = m0 + v + 8 * hf;
    rT[((size_t)row * 8 + h) * 256 + n0 + q] = f2bf(acc.f[v] * sc);
  }
}

// ------------------------------------------------------------- attention core
// one block per (b,q) row. 8 waves. LDS = 48K X-stage + scores + weights + rT
// = 64,768 B (< 64K static budget).
__global__ void __launch_bounds__(256)
k_attention(const float* __restrict__ X0, const unsigned short* __restrict__ rT,
            unsigned short* __restrict__ uOut, float* __restrict__ attnW) {
  __shared__ unsigned short ldsX[24576];   // 48KB: Xr[80][264] or XT[256][96]
  __shared__ float          sS[8 * 168];   // scores  [head][key]
  __shared__ unsigned short sW[16 * 192];  // weights [head(pad16)][key(pad192)]
  __shared__ unsigned short sRT[2048];     // per-row r (8 heads x 256), bf16

  int tid = threadIdx.x, wave = tid >> 5, lane = tid & 31;
  int row = blockIdx.x;
  const float* X = X0 + (size_t)row * (KK * DM);
  int q = lane & 15, hf = lane >> 4;

  // async-stage this row's r into LDS (raw 16B copies, ASYNCcnt-tracked)
  copy16_to_lds(rT + (size_t)row * 2048 + tid * 8, sRT + tid * 8);

  for (int i = tid; i < 16 * 192 / 2; i += 256) ((unsigned*)sW)[i] = 0u;

  // ---- Pass A: scores S[key][h] = X @ r, key-chunks of 80 ----
  for (int cc = 0; cc < 2; ++cc) {
    int c0 = cc * 80;
    __syncthreads();
    {
      int kbase = tid >> 6;            // 0..3
      int c4 = (tid & 63) * 4;
      for (int i = 0; i < 20; ++i) {
        int kl = kbase + 4 * i;        // local key 0..79
        float4 f = *(const float4*)(X + (size_t)(c0 + kl) * DM + c4);
        unsigned* d = (unsigned*)(ldsX + kl * 264 + c4);
        d[0] = pack2(f.x, f.y);
        d[1] = pack2(f.z, f.w);
      }
    }
    if (cc == 0) async_lds_wait();     // rT landed before anyone reads it
    __syncthreads();
    if (wave < 5) {                    // 5 M-tiles of 16 keys
      int m0 = wave * 16;
      const unsigned short* Ab = ldsX + (m0 + q) * 264 + hf * 16;
      const unsigned short* Bb = sRT + q * 256 + hf * 16;
      bool bvalid = (q < 8);           // heads padded 8->16 with zeros
      Acc8 acc; zero8(acc);
      for (int k0 = 0; k0 < 256; k0 += 32) {
        BF16Tile a, b;
        a.u4[0] = *(const uint4*)(Ab + k0);
        a.u4[1] = *(const uint4*)(Ab + k0 + 8);
        if (bvalid) {
          b.u4[0] = *(const uint4*)(Bb + k0);
          b.u4[1] = *(const uint4*)(Bb + k0 + 8);
        } else {
          b.u4[0] = make_uint4(0, 0, 0, 0);
          b.u4[1] = make_uint4(0, 0, 0, 0);
        }
        acc.v = wmma_bf16(a, b, acc.v);
      }
      if (bvalid) {
        float* sp = sS + q * 168 + (c0 + m0 + hf * 8);
#pragma unroll
        for (int v = 0; v < 8; ++v) sp[v] = acc.f[v];
      }
    }
  }
  __syncthreads();

  // ---- softmax over 160 keys, wave == head ----
  {
    int h = wave;
    float e[5]; float mx = -3.0e38f;
#pragma unroll
    for (int j = 0; j < 5; ++j) { e[j] = sS[h * 168 + lane + 32 * j]; mx = fmaxf(mx, e[j]); }
#pragma unroll
    for (int o = 16; o >= 1; o >>= 1) mx = fmaxf(mx, __shfl_xor(mx, o, 32));
    float sum = 0.f;
#pragma unroll
    for (int j = 0; j < 5; ++j) { e[j] = __expf(e[j] - mx); sum += e[j]; }
#pragma unroll
    for (int o = 16; o >= 1; o >>= 1) sum += __shfl_xor(sum, o, 32);
    float inv = 1.f / sum;
    float* ao = attnW + (((size_t)(row / QL) * 8 + h) * QL + (row % QL)) * KK;
#pragma unroll
    for (int j = 0; j < 5; ++j) {
      int k = lane + 32 * j;
      float w = e[j] * inv;
      ao[k] = w;                        // attn_weights output (B,H,Q,K) f32
      sW[h * 192 + k] = f2bf(w);
    }
  }

  // ---- Pass B: u[h][c] = w @ X, chunk-accumulated (X re-read is L2-hot) ----
  Acc8 u0, u1; zero8(u0); zero8(u1);    // wave's N-tiles: wave*32, wave*32+16
  for (int cc = 0; cc < 2; ++cc) {
    int c0 = cc * 80;
    __syncthreads();
    {                                   // zero key-padding cols 80..95
      uint4 z = make_uint4(0, 0, 0, 0);
      unsigned short* rp = ldsX + tid * 96;
      *(uint4*)(rp + 80) = z;
      *(uint4*)(rp + 88) = z;
    }
    {
      int kbase = tid >> 6;
      int c4 = (tid & 63) * 4;
      for (int i = 0; i < 20; ++i) {
        int kl = kbase + 4 * i;
        float4 f = *(const float4*)(X + (size_t)(c0 + kl) * DM + c4);
        ldsX[(c4 + 0) * 96 + kl] = f2bf(f.x);   // transposed stage: XT[c][kl]
        ldsX[(c4 + 1) * 96 + kl] = f2bf(f.y);
        ldsX[(c4 + 2) * 96 + kl] = f2bf(f.z);
        ldsX[(c4 + 3) * 96 + kl] = f2bf(f.w);
      }
    }
    __syncthreads();
    const unsigned short* Ab = sW + q * 192 + c0 + hf * 16;
    const unsigned short* B0 = ldsX + (wave * 32 + q) * 96 + hf * 16;
    const unsigned short* B1 = ldsX + (wave * 32 + 16 + q) * 96 + hf * 16;
    for (int k0 = 0; k0 < 96; k0 += 32) {
      BF16Tile a, b0, b1;
      a.u4[0]  = *(const uint4*)(Ab + k0);
      a.u4[1]  = *(const uint4*)(Ab + k0 + 8);
      b0.u4[0] = *(const uint4*)(B0 + k0);
      b0.u4[1] = *(const uint4*)(B0 + k0 + 8);
      b1.u4[0] = *(const uint4*)(B1 + k0);
      b1.u4[1] = *(const uint4*)(B1 + k0 + 8);
      u0.v = wmma_bf16(a, b0, u0.v);
      u1.v = wmma_bf16(a, b1, u1.v);
    }
  }
  if (hf == 0) {                        // rows 0..7 of the 16-row D tile = heads
    unsigned short* up = uOut + (size_t)row * 2048;
#pragma unroll
    for (int v = 0; v < 8; ++v) {
      up[(size_t)v * 256 + wave * 32 + q]      = f2bf(u0.f[v]);
      up[(size_t)v * 256 + wave * 32 + 16 + q] = f2bf(u1.f[v]);
    }
  }
}

// --------------------------------------------------------------------- launch
extern "C" void kernel_launch(void* const* d_in, const int* in_sizes, int n_in,
                              void* d_out, int out_size, void* d_ws, size_t ws_size,
                              hipStream_t stream) {
  (void)in_sizes; (void)n_in; (void)out_size; (void)ws_size;
  const float* tq  = (const float*)d_in[0];
  const float* sf  = (const float*)d_in[1];
  const float* Wq  = (const float*)d_in[2];
  const float* bq  = (const float*)d_in[3];
  const float* Wk  = (const float*)d_in[4];
  // d_in[5] = bk: softmax-invariant, intentionally unused
  const float* Wv  = (const float*)d_in[6];
  const float* bv  = (const float*)d_in[7];
  const float* Wo  = (const float*)d_in[8];
  const float* bo  = (const float*)d_in[9];
  const float* g1  = (const float*)d_in[10];
  const float* bn1 = (const float*)d_in[11];
  const float* g2  = (const float*)d_in[12];
  const float* bn2 = (const float*)d_in[13];
  const float* W1  = (const float*)d_in[14];
  const float* bf1 = (const float*)d_in[15];
  const float* W2  = (const float*)d_in[16];
  const float* bf2 = (const float*)d_in[17];

  char* ws = (char*)d_ws;
  size_t off = 0;
  auto alloc = [&](size_t bytes) -> char* {
    char* p = ws + off;
    off += (bytes + 255) & ~(size_t)255;
    return p;
  };
  unsigned short* WqT = (unsigned short*)alloc(256 * 256 * 2);
  unsigned short* WkB = (unsigned short*)alloc(256 * 256 * 2);
  unsigned short* W1T = (unsigned short*)alloc(256 * 1024 * 2);
  unsigned short* W2T = (unsigned short*)alloc(1024 * 256 * 2);
  unsigned short* MT  = (unsigned short*)alloc(256 * 2048 * 2);
  float*          c1  = (float*)alloc(256 * 4);
  unsigned short* xnq = (unsigned short*)alloc((size_t)NR * 256 * 2);
  unsigned short* qp  = (unsigned short*)alloc((size_t)NR * 256 * 2);
  unsigned short* rT  = (unsigned short*)alloc((size_t)NR * 2048 * 2);
  unsigned short* uW  = (unsigned short*)alloc((size_t)NR * 2048 * 2);
  float*          xr  = (float*)alloc((size_t)NR * 256 * 4);
  unsigned short* xn2 = (unsigned short*)alloc((size_t)NR * 256 * 2);
  unsigned short* hh  = (unsigned short*)alloc((size_t)NR * 1024 * 2);

  float* outMain = (float*)d_out;                       // (B,Q,256)
  float* outAttn = outMain + (size_t)NR * 256;          // (B,H,Q,K)

  // weight prep (bf16 + transposes to B-operand layout)
  k_convert<<<256, 256, 0, stream>>>(Wq, WqT, 256, 256, 1);
  k_convert<<<256, 256, 0, stream>>>(Wk, WkB, 256, 256, 0);
  k_convert<<<1024, 256, 0, stream>>>(W1, W1T, 256, 1024, 1);
  k_convert<<<1024, 256, 0, stream>>>(W2, W2T, 1024, 256, 1);
  k_mstack<<<2048, 256, 0, stream>>>(Wv, Wo, MT);
  k_c1<<<1, 256, 0, stream>>>(bv, Wo, bo, c1);

  // LN(queries) -> qproj -> per-head fold rT
  k_layernorm<<<360, 256, 0, stream>>>(tq, g1, bn1, xnq);
  k_gemm<<<dim3(180, 1), 256, 0, stream>>>(xnq, 256, WqT, 256, 256, bq,
                                           nullptr, 0, nullptr, qp, 256, 0);
  k_fold<<<dim3(180, 2, 8), 256, 0, stream>>>(qp, WkB, rT);

  // fused attention over sampled_features (single HBM pass)
  k_attention<<<2880, 256, 0, stream>>>(sf, rT, uW, outAttn);

  // x = target + U @ Mstack + c1 ; then FFN
  k_gemm<<<dim3(180, 1), 256, 0, stream>>>(uW, 2048, MT, 2048, 2048, c1,
                                           tq, 256, xr, nullptr, 256, 1);
  k_layernorm<<<360, 256, 0, stream>>>(xr, g2, bn2, xn2);
  k_gemm<<<dim3(180, 4), 256, 0, stream>>>(xn2, 256, W1T, 256, 256, bf1,
                                           nullptr, 0, nullptr, hh, 1024, 2);
  k_gemm<<<dim3(180, 1), 256, 0, stream>>>(hh, 1024, W2T, 1024, 1024, bf2,
                                           xr, 256, outMain, nullptr, 256, 1);
}
